// UltimateExcellencePiMoE_44942537785770
// MI455X (gfx1250) — compile-verified, matrix operand order checked by reading, not run on hardware
//
#include <hip/hip_runtime.h>
#include <hip/hip_bf16.h>
#include <math.h>

// dims
#define BDIM   4
#define SDIM   256
#define DDIM   1024
#define EDIM   2048
#define HHEADS 16
#define HD     128
#define NEXP   4

typedef _Float16 f16;
typedef __attribute__((ext_vector_type(16))) _Float16 v16h;
typedef __attribute__((ext_vector_type(8)))  float    v8f;

// CDNA5 16-bit A/B fragment: lane half (lane>>4) selects K-subgroup.
// element e -> local k within a 32-wide K chunk (e=0..7 and 8..15 contiguous).
__device__ __forceinline__ int frag_k(int e, int half) {
  return (e < 8) ? (half * 8 + e) : (16 + half * 8 + (e - 8));
}

__device__ __forceinline__ v8f wmma16x16x32(v16h a, v16h b, v8f c) {
  // D = A(16x32) * B(32x16) + C, f16 in / f32 accum
  return __builtin_amdgcn_wmma_f32_16x16x32_f16(false, a, false, b, (short)0, c,
                                                false, false);
}

__device__ __forceinline__ float sigmoidf_(float x) {
  return 1.f / (1.f + __expf(-x));
}

// async copy of 16B/lane: global -> LDS, tracked by ASYNCcnt
__device__ __forceinline__ void async_b128(unsigned lds_addr, const float* gp) {
  asm volatile("global_load_async_to_lds_b128 %0, %1, off"
               :: "v"(lds_addr), "v"(gp) : "memory");
}

// ---------------------------------------------------------------------------
// Generic Y = act(A[M,K] @ W[N,K]^T + bias1 (+bias2)), fp32 in/out, f16 WMMA.
// Block: 256 threads (8 waves). Block tile 128(M) x 64(N), K chunked by 32.
// Double-buffered ASYNC staging (global_load_async_to_lds_b128 + ASYNCcnt):
// chunk kb+1 streams into LDS while chunk kb feeds the WMMAs.
// ACT: 0 none, 1 relu, 2 sigmoid, 3 tanh
// ---------------------------------------------------------------------------
template <int ACT>
__global__ __launch_bounds__(256) void gemm_bt_kernel(
    const float* __restrict__ A, const float* __restrict__ W,
    const float* __restrict__ bias1, const float* __restrict__ bias2,
    float* __restrict__ Y, int M, int N, int K) {
  __shared__ float sA[2][128][32];  // 32KB (fp32; convert at fragment build)
  __shared__ float sB[2][64][32];   // 16KB  sB[b][n][k] = W[n0+n][kb+k]
  const int tid  = threadIdx.x;
  const int lane = tid & 31;
  const int wave = tid >> 5;
  const int half = lane >> 4;
  const int l16  = lane & 15;
  const int wm   = wave & 3;   // M group (32 rows)
  const int wn   = wave >> 2;  // N group (32 cols)
  const int m0   = blockIdx.y * 128;
  const int n0   = blockIdx.x * 64;

  const unsigned ldsA = (unsigned)(unsigned long long)(&sA[0][0][0]);
  const unsigned ldsB = (unsigned)(unsigned long long)(&sB[0][0][0]);

  // per-thread async issue for one K-chunk into buffer `buf`:
  // A tile 128x32 fp32 = 1024 x b128 -> 4 per thread; W tile 64x32 -> 2.
  auto issue = [&](int kb, int buf) {
    const int kbase = kb << 5;
#pragma unroll
    for (int q = 0; q < 4; ++q) {
      int cid = tid + q * 256;          // b128 chunk id
      int r = cid >> 3, c = (cid & 7) * 4;
      async_b128(ldsA + (unsigned)((buf * 128 + r) * 32 + c) * 4u,
                 &A[(size_t)(m0 + r) * K + kbase + c]);
    }
#pragma unroll
    for (int q = 0; q < 2; ++q) {
      int cid = tid + q * 256;
      int r = cid >> 3, c = (cid & 7) * 4;
      async_b128(ldsB + (unsigned)((buf * 64 + r) * 32 + c) * 4u,
                 &W[(size_t)(n0 + r) * K + kbase + c]);
    }
  };

  v8f acc[2][2] = {};
  const int KT = K >> 5;
  issue(0, 0);
  for (int kb = 0; kb < KT; ++kb) {
    const int buf = kb & 1;
    if (kb + 1 < KT) {
      issue(kb + 1, buf ^ 1);
      asm volatile("s_wait_asynccnt 0x6" ::: "memory");  // chunk kb landed
    } else {
      asm volatile("s_wait_asynccnt 0x0" ::: "memory");
    }
    __syncthreads();  // all waves' deposits for `buf` visible
    v16h aF[2], bF[2];
#pragma unroll
    for (int i = 0; i < 2; ++i) {
      int row = wm * 32 + i * 16 + l16;
#pragma unroll
      for (int e = 0; e < 16; ++e) aF[i][e] = (f16)sA[buf][row][frag_k(e, half)];
    }
#pragma unroll
    for (int j = 0; j < 2; ++j) {
      int col = wn * 32 + j * 16 + l16;
#pragma unroll
      for (int e = 0; e < 16; ++e) bF[j][e] = (f16)sB[buf][col][frag_k(e, half)];
    }
#pragma unroll
    for (int i = 0; i < 2; ++i)
#pragma unroll
      for (int j = 0; j < 2; ++j) acc[i][j] = wmma16x16x32(aF[i], bF[j], acc[i][j]);
    __syncthreads();  // done reading `buf`; safe to overwrite next iteration
  }
#pragma unroll
  for (int i = 0; i < 2; ++i) {
#pragma unroll
    for (int j = 0; j < 2; ++j) {
      int n = n0 + wn * 32 + j * 16 + l16;
      float bv = bias1 ? bias1[n] : 0.f;
      if (bias2) bv += bias2[n];
#pragma unroll
      for (int r = 0; r < 8; ++r) {
        int m = m0 + wm * 32 + i * 16 + r + 8 * half;  // C/D layout
        float v = acc[i][j][r] + bv;
        if (ACT == 1) v = fmaxf(v, 0.f);
        else if (ACT == 2) v = sigmoidf_(v);
        else if (ACT == 3) v = tanhf(v);
        Y[(size_t)m * N + n] = v;
      }
    }
  }
}

// ---------------------------------------------------------------------------
// Attention: one wave per (b, h, 16-query-row tile). S=256, HD=128.
// Scores in 16 WMMA accumulators, wave-level softmax (xor-shuffle within
// 16-lane halves matches the C-layout row ownership), P staged via LDS to
// A-fragment layout, then P@V with 8 output accumulators.
// ---------------------------------------------------------------------------
__global__ __launch_bounds__(32) void attn_kernel(const float* __restrict__ qkv,
                                                  float* __restrict__ o) {
  __shared__ f16 sP[16][256];  // 8KB
  const int lane = threadIdx.x & 31;
  const int half = lane >> 4;
  const int l16  = lane & 15;
  const int bh   = blockIdx.x >> 4;
  const int qt   = blockIdx.x & 15;
  const int b    = bh / HHEADS;
  const int h    = bh % HHEADS;
  const size_t rs = 3 * EDIM;
  const float* qb = qkv + (size_t)(b * SDIM) * rs + h * HD;
  const float* kb = qb + EDIM;
  const float* vb = qb + 2 * EDIM;

  v16h qF[4];
#pragma unroll
  for (int c = 0; c < 4; ++c) {
    int s = qt * 16 + l16;
#pragma unroll
    for (int e = 0; e < 16; ++e)
      qF[c][e] = (f16)qb[(size_t)s * rs + c * 32 + frag_k(e, half)];
  }

  v8f accS[16] = {};
#pragma unroll
  for (int nt = 0; nt < 16; ++nt) {
#pragma unroll
    for (int c = 0; c < 4; ++c) {
      v16h kF;
      int sk = nt * 16 + l16;
#pragma unroll
      for (int e = 0; e < 16; ++e)
        kF[e] = (f16)kb[(size_t)sk * rs + c * 32 + frag_k(e, half)];
      accS[nt] = wmma16x16x32(qF[c], kF, accS[nt]);
    }
  }

  const float scale = 0.08838834764831845f;  // 1/sqrt(128)
  float mx[8], sm[8];
#pragma unroll
  for (int r = 0; r < 8; ++r) {
    float v = -3.4e38f;
#pragma unroll
    for (int nt = 0; nt < 16; ++nt) v = fmaxf(v, accS[nt][r] * scale);
#pragma unroll
    for (int msk = 8; msk >= 1; msk >>= 1) v = fmaxf(v, __shfl_xor(v, msk, 32));
    mx[r] = v;
  }
#pragma unroll
  for (int r = 0; r < 8; ++r) {
    float s = 0.f;
#pragma unroll
    for (int nt = 0; nt < 16; ++nt) {
      float p = __expf(accS[nt][r] * scale - mx[r]);
      accS[nt][r] = p;
      s += p;
    }
#pragma unroll
    for (int msk = 8; msk >= 1; msk >>= 1) s += __shfl_xor(s, msk, 32);
    sm[r] = s;
  }
#pragma unroll
  for (int nt = 0; nt < 16; ++nt)
#pragma unroll
    for (int r = 0; r < 8; ++r)
      sP[r + 8 * half][nt * 16 + l16] = (f16)(accS[nt][r] / sm[r]);
  __syncthreads();

  v8f accO[8] = {};
#pragma unroll
  for (int c = 0; c < 8; ++c) {
    v16h pF;
#pragma unroll
    for (int e = 0; e < 16; ++e) pF[e] = sP[l16][c * 32 + frag_k(e, half)];
#pragma unroll
    for (int j = 0; j < 8; ++j) {
      v16h vF;
#pragma unroll
      for (int e = 0; e < 16; ++e) {
        int sk = c * 32 + frag_k(e, half);
        vF[e] = (f16)vb[(size_t)sk * rs + j * 16 + l16];  // B non-transposed
      }
      accO[j] = wmma16x16x32(pF, vF, accO[j]);
    }
  }
  float* ob = o + (size_t)(b * SDIM) * EDIM + h * HD;
#pragma unroll
  for (int j = 0; j < 8; ++j)
#pragma unroll
    for (int r = 0; r < 8; ++r) {
      int m = r + 8 * half;
      ob[(size_t)(qt * 16 + m) * EDIM + j * 16 + l16] = accO[j][r];
    }
}

// ---------------------------------------------------------------------------
// LSTM recurrent GEMM: g[16,8192] = hpad[16,2048] @ whh[8192,2048]^T
// One wave per 16x16 output tile (512 tiles); whh stays resident in L2.
// ---------------------------------------------------------------------------
__global__ __launch_bounds__(128) void lstm_gemm_step(
    const float* __restrict__ hpad, const float* __restrict__ whh,
    float* __restrict__ g) {
  const int tid = threadIdx.x;
  const int lane = tid & 31, wave = tid >> 5;
  const int half = lane >> 4, l16 = lane & 15;
  const int n0 = blockIdx.x * 64 + wave * 16;
  v8f acc = {};
  for (int kc = 0; kc < EDIM / 32; ++kc) {
    v16h aF, bF;
#pragma unroll
    for (int e = 0; e < 16; ++e) {
      int k = kc * 32 + frag_k(e, half);
      aF[e] = (f16)hpad[l16 * EDIM + k];
      bF[e] = (f16)whh[(size_t)(n0 + l16) * EDIM + k];
    }
    acc = wmma16x16x32(aF, bF, acc);
  }
#pragma unroll
  for (int r = 0; r < 8; ++r) {
    int m = r + 8 * half;
    g[(size_t)m * (4 * EDIM) + n0 + l16] = acc[r];
  }
}

__global__ void lstm_update(const float* __restrict__ pre,
                            const float* __restrict__ g, float* __restrict__ h,
                            float* __restrict__ c, float* __restrict__ m,
                            int t) {
  int idx = blockIdx.x * blockDim.x + threadIdx.x;
  if (idx >= BDIM * EDIM) return;
  int b = idx / EDIM, u = idx % EDIM;
  size_t pb = (size_t)(b * SDIM + t) * 4 * EDIM;
  size_t gb = (size_t)b * 4 * EDIM;
  float gi = pre[pb + u] + g[gb + u];
  float gf = pre[pb + EDIM + u] + g[gb + EDIM + u];
  float gg = pre[pb + 2 * EDIM + u] + g[gb + 2 * EDIM + u];
  float go = pre[pb + 3 * EDIM + u] + g[gb + 3 * EDIM + u];
  float cn = sigmoidf_(gf) * c[(size_t)b * EDIM + u] + sigmoidf_(gi) * tanhf(gg);
  float hn = sigmoidf_(go) * tanhf(cn);
  c[(size_t)b * EDIM + u] = cn;
  h[(size_t)b * EDIM + u] = hn;
  m[(size_t)(b * SDIM + t) * EDIM + u] = hn;
}

// ---------------------------------------------------------------------------
__global__ __launch_bounds__(256) void layernorm_kernel(
    float* __restrict__ y, const float* __restrict__ g,
    const float* __restrict__ bta, int dim) {
  __shared__ float red[256];
  int row = blockIdx.x, tid = threadIdx.x;
  float* yr = y + (size_t)row * dim;
  float s = 0.f;
  for (int d = tid; d < dim; d += 256) s += yr[d];
  red[tid] = s; __syncthreads();
  for (int o = 128; o > 0; o >>= 1) { if (tid < o) red[tid] += red[tid + o]; __syncthreads(); }
  float mean = red[0] / dim;
  __syncthreads();
  float v = 0.f;
  for (int d = tid; d < dim; d += 256) { float t = yr[d] - mean; v += t * t; }
  red[tid] = v; __syncthreads();
  for (int o = 128; o > 0; o >>= 1) { if (tid < o) red[tid] += red[tid + o]; __syncthreads(); }
  float inv = rsqrtf(red[0] / dim + 1e-5f);
  __syncthreads();
  for (int d = tid; d < dim; d += 256) yr[d] = (yr[d] - mean) * inv * g[d] + bta[d];
}

__global__ __launch_bounds__(256) void ln_accum_kernel(
    const float* __restrict__ src, const float* __restrict__ g,
    const float* __restrict__ bta, const float* __restrict__ wgate, int expert,
    float* __restrict__ out) {
  __shared__ float red[256];
  int row = blockIdx.x, tid = threadIdx.x;  // row = b*S + s
  const float* yr = src + (size_t)row * DDIM;
  float s = 0.f;
  for (int d = tid; d < DDIM; d += 256) s += yr[d];
  red[tid] = s; __syncthreads();
  for (int o = 128; o > 0; o >>= 1) { if (tid < o) red[tid] += red[tid + o]; __syncthreads(); }
  float mean = red[0] / DDIM;
  __syncthreads();
  float v = 0.f;
  for (int d = tid; d < DDIM; d += 256) { float t = yr[d] - mean; v += t * t; }
  red[tid] = v; __syncthreads();
  for (int o = 128; o > 0; o >>= 1) { if (tid < o) red[tid] += red[tid + o]; __syncthreads(); }
  float inv = rsqrtf(red[0] / DDIM + 1e-5f);
  float w = wgate[(row / SDIM) * NEXP + expert];
  for (int d = tid; d < DDIM; d += 256)
    out[(size_t)row * DDIM + d] += w * ((yr[d] - mean) * inv * g[d] + bta[d]);
}

// ---------------------------------------------------------------------------
__global__ __launch_bounds__(256) void gate_kernel(
    const float* __restrict__ x, const float* __restrict__ gw,
    const float* __restrict__ gb, float* __restrict__ wgate) {
  __shared__ float sp[DDIM];
  __shared__ float red[256 * NEXP];
  int b = blockIdx.x, tid = threadIdx.x;
  for (int d = tid; d < DDIM; d += 256) {
    float s = 0.f;
    for (int ss = 0; ss < SDIM; ++ss) s += x[(size_t)(b * SDIM + ss) * DDIM + d];
    sp[d] = s / (float)SDIM;
  }
  __syncthreads();
  float p[NEXP] = {};
  for (int d = tid; d < DDIM; d += 256) {
    float pv = sp[d];
    for (int e = 0; e < NEXP; ++e) p[e] += pv * gw[d * NEXP + e];
  }
  for (int e = 0; e < NEXP; ++e) red[tid * NEXP + e] = p[e];
  __syncthreads();
  for (int o = 128; o > 0; o >>= 1) {
    if (tid < o)
      for (int e = 0; e < NEXP; ++e) red[tid * NEXP + e] += red[(tid + o) * NEXP + e];
    __syncthreads();
  }
  if (tid == 0) {
    float lg[NEXP], mx = -3.4e38f;
    for (int e = 0; e < NEXP; ++e) { lg[e] = red[e] + gb[e]; mx = fmaxf(mx, lg[e]); }
    float sum = 0.f, pr[NEXP];
    for (int e = 0; e < NEXP; ++e) { pr[e] = __expf(lg[e] - mx); sum += pr[e]; }
    for (int e = 0; e < NEXP; ++e) pr[e] /= sum;
    int i1 = 0;
    for (int e = 1; e < NEXP; ++e) if (pr[e] > pr[i1]) i1 = e;
    int i2 = -1;
    for (int e = 0; e < NEXP; ++e) if (e != i1 && (i2 < 0 || pr[e] > pr[i2])) i2 = e;
    float tot = pr[i1] + pr[i2];
    for (int e = 0; e < NEXP; ++e) wgate[b * NEXP + e] = 0.f;
    wgate[b * NEXP + i1] = pr[i1] / tot;
    wgate[b * NEXP + i2] = pr[i2] / tot;
  }
}

__global__ void zero_kernel(float* __restrict__ p, size_t n) {
  size_t i = (size_t)blockIdx.x * blockDim.x + threadIdx.x;
  if (i < n) p[i] = 0.f;
}

__global__ void mul_kernel(const float* __restrict__ a,
                           const float* __restrict__ b, float* __restrict__ o,
                           size_t n) {
  size_t i = (size_t)blockIdx.x * blockDim.x + threadIdx.x;
  if (i < n) o[i] = a[i] * b[i];
}

// ---------------------------------------------------------------------------
extern "C" void kernel_launch(void* const* d_in, const int* in_sizes, int n_in,
                              void* d_out, int out_size, void* d_ws,
                              size_t ws_size, hipStream_t stream) {
  const float* x        = (const float*)d_in[0];
  const float* gate_w   = (const float*)d_in[1];
  const float* gate_b   = (const float*)d_in[2];
  const float* enc_w1   = (const float*)d_in[3];
  const float* enc_b1   = (const float*)d_in[4];
  const float* enc_w2   = (const float*)d_in[5];
  const float* enc_b2   = (const float*)d_in[6];
  const float* enc_ln_g = (const float*)d_in[7];
  const float* enc_ln_b = (const float*)d_in[8];
  const float* ain_w    = (const float*)d_in[9];
  const float* ain_b    = (const float*)d_in[10];
  const float* aout_w   = (const float*)d_in[11];
  const float* aout_b   = (const float*)d_in[12];
  const float* w_ih     = (const float*)d_in[13];
  const float* w_hh     = (const float*)d_in[14];
  const float* b_ih     = (const float*)d_in[15];
  const float* b_hh     = (const float*)d_in[16];
  const float* exc_w1   = (const float*)d_in[17];
  const float* exc_b1   = (const float*)d_in[18];
  const float* exc_w2   = (const float*)d_in[19];
  const float* exc_b2   = (const float*)d_in[20];
  const float* opt_w1   = (const float*)d_in[21];
  const float* opt_b1   = (const float*)d_in[22];
  const float* opt_w2   = (const float*)d_in[23];
  const float* opt_b2   = (const float*)d_in[24];
  const float* dec_w1   = (const float*)d_in[25];
  const float* dec_b1   = (const float*)d_in[26];
  const float* dec_w2   = (const float*)d_in[27];
  const float* dec_b2   = (const float*)d_in[28];
  const float* dec_ln_g = (const float*)d_in[29];
  const float* dec_ln_b = (const float*)d_in[30];
  float* out = (float*)d_out;

  float* ws = (float*)d_ws;
  size_t off = 0;
  auto carve = [&](size_t n) { float* p = ws + off; off += n; return p; };
  float* wgate = carve(BDIM * NEXP);
  float* bufA  = carve((size_t)1024 * EDIM);
  float* bufB  = carve((size_t)1024 * EDIM);
  float* bufC  = carve((size_t)1024 * EDIM);
  float* qkvb  = carve((size_t)1024 * 3 * EDIM);
  float* preb  = carve((size_t)1024 * 4 * EDIM);
  float* gbuf  = carve((size_t)16 * 4 * EDIM);
  float* hbuf  = carve((size_t)16 * EDIM);
  float* cbuf  = carve((size_t)16 * EDIM);
  float* mbuf  = carve((size_t)1024 * EDIM);

  const int M = BDIM * SDIM;  // 1024 tokens
  zero_kernel<<<(M * DDIM + 255) / 256, 256, 0, stream>>>(out, (size_t)M * DDIM);
  gate_kernel<<<BDIM, 256, 0, stream>>>(x, gate_w, gate_b, wgate);

  for (int e = 0; e < NEXP; ++e) {
    const float* ew1  = enc_w1 + (size_t)e * EDIM * DDIM;
    const float* eb1  = enc_b1 + (size_t)e * EDIM;
    const float* ew2  = enc_w2 + (size_t)e * EDIM * EDIM;
    const float* eb2  = enc_b2 + (size_t)e * EDIM;
    const float* elg  = enc_ln_g + (size_t)e * EDIM;
    const float* elb  = enc_ln_b + (size_t)e * EDIM;
    const float* aiw  = ain_w + (size_t)e * 3 * EDIM * EDIM;
    const float* aib  = ain_b + (size_t)e * 3 * EDIM;
    const float* aow  = aout_w + (size_t)e * EDIM * EDIM;
    const float* aob  = aout_b + (size_t)e * EDIM;
    const float* wih  = w_ih + (size_t)e * 4 * EDIM * EDIM;
    const float* whh  = w_hh + (size_t)e * 4 * EDIM * EDIM;
    const float* bih  = b_ih + (size_t)e * 4 * EDIM;
    const float* bhh  = b_hh + (size_t)e * 4 * EDIM;
    const float* xw1  = exc_w1 + (size_t)e * EDIM * EDIM;
    const float* xb1  = exc_b1 + (size_t)e * EDIM;
    const float* xw2  = exc_w2 + (size_t)e * EDIM * EDIM;
    const float* xb2  = exc_b2 + (size_t)e * EDIM;
    const float* ow1  = opt_w1 + (size_t)e * EDIM * EDIM;
    const float* ob1  = opt_b1 + (size_t)e * EDIM;
    const float* ow2  = opt_w2 + (size_t)e * EDIM * EDIM;
    const float* ob2  = opt_b2 + (size_t)e * EDIM;
    const float* dw1  = dec_w1 + (size_t)e * EDIM * EDIM;
    const float* db1  = dec_b1 + (size_t)e * EDIM;
    const float* dw2  = dec_w2 + (size_t)e * DDIM * EDIM;
    const float* db2  = dec_b2 + (size_t)e * DDIM;
    const float* dlg  = dec_ln_g + (size_t)e * DDIM;
    const float* dlb  = dec_ln_b + (size_t)e * DDIM;

    // encoder: x -> relu -> linear -> LN
    gemm_bt_kernel<1><<<dim3(EDIM / 64, M / 128), 256, 0, stream>>>(
        x, ew1, eb1, nullptr, bufA, M, EDIM, DDIM);
    gemm_bt_kernel<0><<<dim3(EDIM / 64, M / 128), 256, 0, stream>>>(
        bufA, ew2, eb2, nullptr, bufB, M, EDIM, EDIM);
    layernorm_kernel<<<M, 256, 0, stream>>>(bufB, elg, elb, EDIM);

    // attention
    gemm_bt_kernel<0><<<dim3(3 * EDIM / 64, M / 128), 256, 0, stream>>>(
        bufB, aiw, aib, nullptr, qkvb, M, 3 * EDIM, EDIM);
    attn_kernel<<<BDIM * HHEADS * (SDIM / 16), 32, 0, stream>>>(qkvb, bufA);
    gemm_bt_kernel<0><<<dim3(EDIM / 64, M / 128), 256, 0, stream>>>(
        bufA, aow, aob, nullptr, bufB, M, EDIM, EDIM);

    // LSTM input projection (bias = b_ih + b_hh), then 256 recurrent steps
    gemm_bt_kernel<0><<<dim3(4 * EDIM / 64, M / 128), 256, 0, stream>>>(
        bufB, wih, bih, bhh, preb, M, 4 * EDIM, EDIM);
    zero_kernel<<<(16 * EDIM + 255) / 256, 256, 0, stream>>>(hbuf, 16 * EDIM);
    zero_kernel<<<(16 * EDIM + 255) / 256, 256, 0, stream>>>(cbuf, 16 * EDIM);
    for (int t = 0; t < SDIM; ++t) {
      lstm_gemm_step<<<4 * EDIM / 64, 128, 0, stream>>>(hbuf, whh, gbuf);
      lstm_update<<<(BDIM * EDIM + 255) / 256, 256, 0, stream>>>(
          preb, gbuf, hbuf, cbuf, mbuf, t);
    }

    // excellence / optimality heads
    gemm_bt_kernel<1><<<dim3(EDIM / 64, M / 128), 256, 0, stream>>>(
        mbuf, xw1, xb1, nullptr, bufA, M, EDIM, EDIM);
    gemm_bt_kernel<2><<<dim3(EDIM / 64, M / 128), 256, 0, stream>>>(
        bufA, xw2, xb2, nullptr, bufB, M, EDIM, EDIM);  // sigmoid
    gemm_bt_kernel<1><<<dim3(EDIM / 64, M / 128), 256, 0, stream>>>(
        mbuf, ow1, ob1, nullptr, bufA, M, EDIM, EDIM);
    gemm_bt_kernel<3><<<dim3(EDIM / 64, M / 128), 256, 0, stream>>>(
        bufA, ow2, ob2, nullptr, bufC, M, EDIM, EDIM);  // tanh
    mul_kernel<<<(M * EDIM + 255) / 256, 256, 0, stream>>>(
        bufB, bufC, bufA, (size_t)M * EDIM);

    // decoder + LN + gated accumulate
    gemm_bt_kernel<1><<<dim3(EDIM / 64, M / 128), 256, 0, stream>>>(
        bufA, dw1, db1, nullptr, bufB, M, EDIM, EDIM);
    gemm_bt_kernel<0><<<dim3(DDIM / 64, M / 128), 256, 0, stream>>>(
        bufB, dw2, db2, nullptr, bufC, M, DDIM, EDIM);
    ln_accum_kernel<<<M, 256, 0, stream>>>(bufC, dlg, dlb, wgate, e, out);
  }
}
